// ConvExponential_30202210025802
// MI455X (gfx1250) — compile-verified, matrix operand order checked by reading, not run on hardware
//
#include <hip/hip_runtime.h>

// ---------------------------------------------------------------------------
// ConvExponential for MI455X (gfx1250, wave32, WMMA 16x16x32 f16 -> f32).
//
// ki_new[:,:,u,v] = (1/i) * sum_{p,q} K[:,:,p,q] @ ki_old[:,:,u-p,v-q]
// kg accumulates each ki centered in an 11x11 spatial grid (= d_out).
//
// All matrices live in WMMA-fragment-native layout in workspace:
//   frag(mat, tile, ktile, lane)[16 halves] -> 1KB contiguous per fragment.
// fp32 precision is recovered with an hi/lo f16 split (3 WMMAs per K-step).
// A-fragments are staged global->LDS with GLOBAL_LOAD_ASYNC_TO_LDS_B128
// (ASYNCcnt), double-buffered in 32KB chunks of 4 k-steps.
// ---------------------------------------------------------------------------

typedef __attribute__((ext_vector_type(16))) _Float16 v16h;
typedef __attribute__((ext_vector_type(8)))  _Float16 v8h;
typedef __attribute__((ext_vector_type(8)))  float    v8f;

#define CDIM   512
#define KTILES 16          // 512 / 32
#define KKTAPS 9           // 3x3 taps
#define MAT_HALVES (512 * 512)
// LDS chunk: 4 ktiles x 4 mtiles x {hi,lo} fragments x 512 halves = 16384 halves
#define CHUNK_HALVES 16384

// ---------------------------------------------------------------------------
// d_out = kg seed: zero everywhere, kernel + identity at the centered 3x3.
// ---------------------------------------------------------------------------
__global__ void init_out_kernel(const float* __restrict__ kern,
                                float* __restrict__ out) {
  unsigned tid = blockIdx.x * blockDim.x + threadIdx.x;   // exact grid
  int x = tid % 11;
  int y = (tid / 11) % 11;
  int b = (tid / 121) % 512;
  int a = tid / (121 * 512);
  float v = 0.0f;
  if (y >= 4 && y <= 6 && x >= 4 && x <= 6) {
    v = kern[((size_t)(a * 512 + b)) * 9 + (y - 4) * 3 + (x - 4)];
    if (a == b && y == 5 && x == 5) v += 1.0f;
  }
  out[tid] = v;
}

// ---------------------------------------------------------------------------
// Pack kernel taps into A-fragment layout (hi/lo f16 split).
// A frag slot (t, mt, kt, lane): lane holds A[o = mt*16 + (lane&15)]
//                                          [c = kt*32 + (lane>>4)*16 + h]
// ---------------------------------------------------------------------------
__global__ void pack_A_kernel(const float* __restrict__ kern,
                              _Float16* __restrict__ ahi,
                              _Float16* __restrict__ alo) {
  unsigned tid = blockIdx.x * blockDim.x + threadIdx.x;
  if (tid >= KKTAPS * 32 * 16 * 32) return;
  int lane = tid & 31;
  int kt   = (tid >> 5) & 15;
  int mt   = (tid >> 9) & 31;
  int t    = tid >> 14;
  int o  = mt * 16 + (lane & 15);
  int c0 = kt * 32 + (lane >> 4) * 16;
  v16h hi, lo;
#pragma unroll
  for (int h = 0; h < 16; ++h) {
    float x = kern[((size_t)o * 512 + (c0 + h)) * 9 + t];
    _Float16 xh = (_Float16)x;
    hi[h] = xh;
    lo[h] = (_Float16)(x - (float)xh);
  }
  size_t base = (size_t)tid * 16;
  *(v16h*)(ahi + base) = hi;
  *(v16h*)(alo + base) = lo;
}

// ---------------------------------------------------------------------------
// Pack kernel as the initial ki (B-fragment layout, spatial = the 9 taps).
// B frag slot (s, nt, kt, lane): lane holds B[k = kt*32 + (lane>>4)*16 + h]
//                                          [n = nt*16 + (lane&15)]
// ---------------------------------------------------------------------------
__global__ void pack_B0_kernel(const float* __restrict__ kern,
                               _Float16* __restrict__ bhi,
                               _Float16* __restrict__ blo) {
  unsigned tid = blockIdx.x * blockDim.x + threadIdx.x;
  if (tid >= KKTAPS * 32 * 16 * 32) return;
  int lane = tid & 31;
  int kt   = (tid >> 5) & 15;
  int nt   = (tid >> 9) & 31;
  int s    = tid >> 14;
  int n  = nt * 16 + (lane & 15);
  int k0 = kt * 32 + (lane >> 4) * 16;
  v16h hi, lo;
#pragma unroll
  for (int h = 0; h < 16; ++h) {
    float x = kern[((size_t)(k0 + h) * 512 + n) * 9 + s];
    _Float16 xh = (_Float16)x;
    hi[h] = xh;
    lo[h] = (_Float16)(x - (float)xh);
  }
  size_t base = (size_t)tid * 16;
  *(v16h*)(bhi + base) = hi;
  *(v16h*)(blo + base) = lo;
}

// ---------------------------------------------------------------------------
// Issue async global->LDS copies for one 32KB A-fragment chunk:
// 4 ktiles (kt0..kt0+3) x 4 block m-tiles x {hi,lo}, 16B per lane per issue.
// Fragment (kc, mt, hl) -> LDS halves [((kc*4+mt)*2+hl)*512, +512).
// ---------------------------------------------------------------------------
__device__ __forceinline__ void issue_a_chunk(const _Float16* __restrict__ ahi,
                                              const _Float16* __restrict__ alo,
                                              _Float16* smem, int buf,
                                              size_t abase, int mtb, int kt0,
                                              int tid) {
#pragma unroll
  for (int j = 0; j < 16; ++j) {
    int e    = j * 128 + tid;        // 16-byte unit index within chunk
    int unit = e & 63;               // unit within fragment (64 x 16B = 1KB)
    int frag = e >> 6;
    int hl   = frag & 1;
    int mt   = (frag >> 1) & 3;
    int kc   = frag >> 3;
    const _Float16* src = (hl ? alo : ahi) + abase
        + (((size_t)(mtb + mt) * 16 + (kt0 + kc)) * 32) * 16 + (size_t)unit * 8;
    unsigned dst = (unsigned)(size_t)(smem + (size_t)buf * CHUNK_HALVES
                                      + (size_t)e * 8);
    asm volatile("global_load_async_to_lds_b128 %0, %1, off"
                 :: "v"(dst), "v"(src)
                 : "memory");
  }
}

// ---------------------------------------------------------------------------
// Batched GEMM: one output spatial position per blockIdx.x, 64x64 channel tile
// per block (4 wave32s, 32x32 per wave = 2x2 WMMA C tiles).
// A from LDS (async-staged, double buffered), B direct from global.
// ---------------------------------------------------------------------------
__global__ __launch_bounds__(128)
void exp_gemm_kernel(const _Float16* __restrict__ ahi,
                     const _Float16* __restrict__ alo,
                     const _Float16* __restrict__ bhi,
                     const _Float16* __restrict__ blo,
                     _Float16* __restrict__ ohi,
                     _Float16* __restrict__ olo,
                     float* __restrict__ kg,
                     int S_in, int S_out, float inv, int off, int writeB) {
  __shared__ _Float16 smem[2 * CHUNK_HALVES];   // 64 KB, double buffered

  const int tid  = threadIdx.x;
  const int s    = blockIdx.x;              // output spatial position
  const int u    = s / S_out;
  const int v    = s % S_out;
  const int lane = tid & 31;
  const int w    = tid >> 5;
  const int mtb    = blockIdx.y * 4;                  // block's first m-tile
  const int ntile0 = blockIdx.z * 4 + (w & 1) * 2;    // wave's first n-tile
  const int mwl    = (w >> 1) * 2;                    // wave's m-tile (local)

  v8f acc[2][2] = {};

  for (int p = 0; p < 3; ++p) {
    const int su = u - p;
    if (su < 0 || su >= S_in) continue;
    for (int q = 0; q < 3; ++q) {
      const int sv = v - q;
      if (sv < 0 || sv >= S_in) continue;
      const int t   = p * 3 + q;           // tap -> A matrix
      const int sin = su * S_in + sv;      // input spatial -> B matrix
      const size_t abase = (size_t)t * MAT_HALVES;
      const size_t bbase = (size_t)sin * MAT_HALVES;

      // stage first chunk of this tap's A fragments
      issue_a_chunk(ahi, alo, smem, 0, abase, mtb, 0, tid);

      for (int c = 0; c < 4; ++c) {          // 4 chunks x 4 k-steps
        asm volatile("s_wait_asynccnt 0x0" ::: "memory");
        __syncthreads();                     // chunk c resident for all waves
        if (c < 3)
          issue_a_chunk(ahi, alo, smem, (c + 1) & 1, abase, mtb, (c + 1) * 4, tid);
        const _Float16* sbuf = smem + (size_t)(c & 1) * CHUNK_HALVES;

#pragma unroll
        for (int kc = 0; kc < 4; ++kc) {
          const int kt = c * 4 + kc;
          v16h afh[2], afl[2], bfh[2], bfl[2];
#pragma unroll
          for (int mi = 0; mi < 2; ++mi) {
            const _Float16* sa =
                sbuf + (size_t)((kc * 4 + (mwl + mi)) * 2) * 512 + lane * 16;
            afh[mi] = *(const v16h*)sa;          // ds_load_b128 x2
            afl[mi] = *(const v16h*)(sa + 512);
          }
#pragma unroll
          for (int ni = 0; ni < 2; ++ni) {
            size_t bi = bbase + (((size_t)(ntile0 + ni) * 16 + kt) * 32 + lane) * 16;
            bfh[ni] = *(const v16h*)(bhi + bi);
            bfl[ni] = *(const v16h*)(blo + bi);
          }
#pragma unroll
          for (int mi = 0; mi < 2; ++mi)
#pragma unroll
            for (int ni = 0; ni < 2; ++ni) {
              acc[mi][ni] = __builtin_amdgcn_wmma_f32_16x16x32_f16(
                  false, afh[mi], false, bfh[ni], (short)0, acc[mi][ni], false, false);
              acc[mi][ni] = __builtin_amdgcn_wmma_f32_16x16x32_f16(
                  false, afh[mi], false, bfl[ni], (short)0, acc[mi][ni], false, false);
              acc[mi][ni] = __builtin_amdgcn_wmma_f32_16x16x32_f16(
                  false, afl[mi], false, bfh[ni], (short)0, acc[mi][ni], false, false);
            }
        }
      }
    }
  }

  // Epilogue: scale by 1/i, accumulate into kg (= d_out) at centered offset,
  // and emit next iteration's B fragments (hi/lo f16).
  const int yy = off + u;
  const int xx = off + v;
#pragma unroll
  for (int mi = 0; mi < 2; ++mi) {
#pragma unroll
    for (int ni = 0; ni < 2; ++ni) {
      const int mt = mtb + mwl + mi;
      const int nt = ntile0 + ni;
      const int bcol  = nt * 16 + (lane & 15);
      const int arow0 = mt * 16 + 8 * (lane >> 4);
      v8h hi8, lo8;
#pragma unroll
      for (int r = 0; r < 8; ++r) {
        float d = acc[mi][ni][r] * inv;
        size_t oidx = ((size_t)((arow0 + r) * 512 + bcol)) * 121
                      + (size_t)yy * 11 + xx;
        kg[oidx] += d;
        _Float16 dh = (_Float16)d;
        hi8[r] = dh;
        lo8[r] = (_Float16)(d - (float)dh);
      }
      if (writeB) {
        const int ktp = mt >> 1;
        const int lp  = ((mt & 1) << 4) + (lane & 15);
        size_t bo = ((size_t)s * MAT_HALVES)
                  + (((size_t)nt * 16 + ktp) * 32 + lp) * 16
                  + (lane >> 4) * 8;
        *(v8h*)(ohi + bo) = hi8;
        *(v8h*)(olo + bo) = lo8;
      }
    }
  }
}

// ---------------------------------------------------------------------------
// Host orchestration.
// Workspace (halves): ahi|alo [9*512*512 each],
//   b1hi|b1lo [81*512*512 each] (5x5 and 9x9 ki),
//   b2hi|b2lo [49*512*512 each] (initial 3x3 and 7x7 ki).
// Total = 145,752,064 bytes.
// ---------------------------------------------------------------------------
extern "C" void kernel_launch(void* const* d_in, const int* in_sizes, int n_in,
                              void* d_out, int out_size, void* d_ws, size_t ws_size,
                              hipStream_t stream) {
  const float* kern = (const float*)d_in[0];   // (512,512,3,3) fp32
  float* out = (float*)d_out;                  // (512,512,11,11) fp32

  const size_t A_H  = (size_t)KKTAPS * MAT_HALVES;
  const size_t B1_H = (size_t)81 * MAT_HALVES;
  const size_t B2_H = (size_t)49 * MAT_HALVES;

  _Float16* ws   = (_Float16*)d_ws;
  _Float16* ahi  = ws;
  _Float16* alo  = ahi + A_H;
  _Float16* b1hi = alo + A_H;
  _Float16* b1lo = b1hi + B1_H;
  _Float16* b2hi = b1lo + B1_H;
  _Float16* b2lo = b2hi + B2_H;

  // 1) seed kg in d_out: kernel + identity at centered 3x3.
  init_out_kernel<<<(512u * 512u * 121u) / 256u, 256, 0, stream>>>(kern, out);

  // 2) pack A (9 taps) and initial B (= kernel, 3x3 spatial) fragments.
  pack_A_kernel <<<576, 256, 0, stream>>>(kern, ahi, alo);
  pack_B0_kernel<<<576, 256, 0, stream>>>(kern, b2hi, b2lo);

  // 3) Taylor iterations: ki_i = (K (*) ki_{i-1}) / i ; kg += ki_i (centered).
  exp_gemm_kernel<<<dim3(25, 8, 8), 128, 0, stream>>>(
      ahi, alo, b2hi, b2lo, b1hi, b1lo, out, 3, 5, 0.5f, 3, 1);
  exp_gemm_kernel<<<dim3(49, 8, 8), 128, 0, stream>>>(
      ahi, alo, b1hi, b1lo, b2hi, b2lo, out, 5, 7, 1.0f / 3.0f, 2, 1);
  exp_gemm_kernel<<<dim3(81, 8, 8), 128, 0, stream>>>(
      ahi, alo, b2hi, b2lo, b1hi, b1lo, out, 7, 9, 0.25f, 1, 1);
  exp_gemm_kernel<<<dim3(121, 8, 8), 128, 0, stream>>>(
      ahi, alo, b1hi, b1lo, b2hi, b2lo, out, 9, 11, 0.2f, 0, 0);
}
